// Attention_40699110097460
// MI455X (gfx1250) — compile-verified, matrix operand order checked by reading, not run on hardware
//
#include <hip/hip_runtime.h>
#include <hip/hip_bf16.h>

#define B_ 4
#define S_ 2048
#define D_ 1024
#define H_ 16
#define A_ 64

typedef __attribute__((ext_vector_type(16))) _Float16 v16h;
typedef __attribute__((ext_vector_type(8)))  _Float16 v8h;
typedef __attribute__((ext_vector_type(8)))  float    v8f;

static __device__ inline v16h combine16(v8h lo, v8h hi) {
  v16h r;
#pragma unroll
  for (int i = 0; i < 8; ++i) { r[i] = lo[i]; r[8 + i] = hi[i]; }
  return r;
}

static __device__ inline v8f zero8f() {
  v8f z = {0.f, 0.f, 0.f, 0.f, 0.f, 0.f, 0.f, 0.f};
  return z;
}

// Async global->LDS copy of 16 bytes (GLOBAL_LOAD_ASYNC_TO_LDS_B128, ASYNCcnt).
// ldsAddr: wave-relative LDS byte offset (low 32 bits of a generic LDS pointer).
static __device__ inline void async_copy_b128(unsigned ldsAddr, const void* gptr) {
  unsigned long long ga = (unsigned long long)(uintptr_t)gptr;
  asm volatile("global_load_async_to_lds_b128 %0, %1, off"
               :: "v"(ldsAddr), "v"(ga) : "memory");
}

static __device__ inline void wait_async0() {
  asm volatile("s_wait_asynccnt 0x0" ::: "memory");
}

// ---------------------------------------------------------------------------
// Kernel 1: per-head projection GEMM  y[bh][s][a] = x[b][s][:] @ W[h][:,:] + b[h][:]
// grid = (S/128, B*H, 3), block = 256 (8 waves); each wave does a 16x64 tile.
// ---------------------------------------------------------------------------
__global__ __launch_bounds__(256) void proj_kernel(
    const float* __restrict__ q, const float* __restrict__ k, const float* __restrict__ v,
    const float* __restrict__ Wq, const float* __restrict__ Wk, const float* __restrict__ Wv,
    const float* __restrict__ bq, const float* __restrict__ bk, const float* __restrict__ bv,
    _Float16* __restrict__ qh, _Float16* __restrict__ kh, _Float16* __restrict__ vh)
{
  const int mat = blockIdx.z;
  const float* x    = (mat == 0) ? q  : (mat == 1) ? k  : v;
  const float* W    = (mat == 0) ? Wq : (mat == 1) ? Wk : Wv;
  const float* bias = (mat == 0) ? bq : (mat == 1) ? bk : bv;
  _Float16*    y    = (mat == 0) ? qh : (mat == 1) ? kh : vh;

  const int bh = blockIdx.y;          // 0..B*H-1
  const int b  = bh / H_;
  const int h  = bh % H_;
  const int s0 = blockIdx.x * 128;

  const int tid   = threadIdx.x;
  const int wave  = tid >> 5;
  const int lane  = tid & 31;
  const int lrow  = lane & 15;        // row (A frag) / column (B,C frags)
  const int khalf = lane >> 4;        // which K-half this lane holds

  __shared__ _Float16 Wt[64][32];     // W slice transposed: Wt[a][k_local]

  const float* xbase = x + ((size_t)b * S_ + (size_t)(s0 + wave * 16 + lrow)) * D_;
  const float* Wh    = W + (size_t)h * D_ * A_;

  v8f acc[4];
#pragma unroll
  for (int t = 0; t < 4; ++t) acc[t] = zero8f();

  for (int k0 = 0; k0 < D_; k0 += 32) {
    // cooperative stage of W[k0..k0+31][0..63] (f32) -> Wt[a][k] (f16, transposed)
    {
      const int kl = tid >> 3;                 // 0..31
      const int a0 = (tid & 7) * 8;            // 0..56
      v8f wv = *(const v8f*)(Wh + (size_t)(k0 + kl) * A_ + a0);
#pragma unroll
      for (int i = 0; i < 8; ++i) Wt[a0 + i][kl] = (_Float16)wv[i];
    }
    __syncthreads();

    // A fragment: 16 rows x 32 K of x, converted f32 -> f16
    v8f xlo = *(const v8f*)(xbase + k0 + khalf * 8);
    v8f xhi = *(const v8f*)(xbase + k0 + 16 + khalf * 8);
    v16h afrag;
#pragma unroll
    for (int i = 0; i < 8; ++i) { afrag[i] = (_Float16)xlo[i]; afrag[8 + i] = (_Float16)xhi[i]; }

#pragma unroll
    for (int t = 0; t < 4; ++t) {
      const _Float16* wl = &Wt[lrow + 16 * t][0];        // lane's output column
      v8h blo = *(const v8h*)(wl + khalf * 8);
      v8h bhi = *(const v8h*)(wl + 16 + khalf * 8);
      v16h bfrag = combine16(blo, bhi);
      acc[t] = __builtin_amdgcn_wmma_f32_16x16x32_f16(
          false, afrag, false, bfrag, (short)0, acc[t], false, false);
    }
    __syncthreads();
  }

  // epilogue: add bias, convert to f16, store y[bh][s][a]
  _Float16* ybase = y + ((size_t)bh * S_ + (size_t)(s0 + wave * 16)) * A_;
#pragma unroll
  for (int t = 0; t < 4; ++t) {
    const int n = lrow + 16 * t;
    const float bb = bias[h * A_ + n];
#pragma unroll
    for (int r = 0; r < 8; ++r) {
      const int row = r + 8 * khalf;
      ybase[(size_t)row * A_ + n] = (_Float16)(acc[t][r] + bb);
    }
  }
}

// ---------------------------------------------------------------------------
// Kernel 2: fused flash attention per (b,h). grid = (S/128, B*H), block = 256.
// Each wave owns 16 query rows; KV walked in 64-wide chunks with online softmax.
// K chunk staged via async global->LDS (ASYNCcnt); V staged transposed via VGPRs.
// ---------------------------------------------------------------------------
__global__ __launch_bounds__(256) void attn_kernel(
    const _Float16* __restrict__ qh, const _Float16* __restrict__ kh,
    const _Float16* __restrict__ vh, float* __restrict__ out)
{
  const int bh = blockIdx.y;
  const int b  = bh / H_;
  const int h  = bh % H_;
  const int s0 = blockIdx.x * 128;

  const int tid   = threadIdx.x;
  const int wave  = tid >> 5;
  const int lane  = tid & 31;
  const int lrow  = lane & 15;
  const int khalf = lane >> 4;

  __shared__ _Float16 Kt[64][64];       // K chunk, [skv_local][a] (row major)
  __shared__ _Float16 Vt[64][64];       // V chunk transposed, [a][skv_local]
  __shared__ _Float16 Pl[8][16 * 64];   // per-wave P relayout scratch

  // Q fragments for this wave's 16 rows (K dim = A = 64 -> two 16x32 frags)
  const _Float16* qrow = qh + ((size_t)bh * S_ + (size_t)(s0 + wave * 16 + lrow)) * A_;
  v16h qf[2];
#pragma unroll
  for (int ka = 0; ka < 2; ++ka) {
    v8h lo = *(const v8h*)(qrow + ka * 32 + khalf * 8);
    v8h hi = *(const v8h*)(qrow + ka * 32 + 16 + khalf * 8);
    qf[ka] = combine16(lo, hi);
  }

  v8f acc[4];
#pragma unroll
  for (int t = 0; t < 4; ++t) acc[t] = zero8f();
  float mrow[8], lsum[8];
#pragma unroll
  for (int r = 0; r < 8; ++r) { mrow[r] = -1e30f; lsum[r] = 0.0f; }

  const _Float16* kbase = kh + (size_t)bh * S_ * A_;
  const _Float16* vbase = vh + (size_t)bh * S_ * A_;

  // staging coordinates: thread covers row kl, 16 columns starting at a0
  const int skl = tid >> 2;            // 0..63
  const int sa0 = (tid & 3) * 16;      // 0,16,32,48

  for (int j0 = 0; j0 < S_; j0 += 64) {
    // --- K chunk: async global -> LDS (row-major copy, no VGPR round trip)
    {
      const _Float16* kg = kbase + (size_t)(j0 + skl) * A_ + sa0;
      unsigned laddr = (unsigned)(uintptr_t)&Kt[skl][sa0];
      async_copy_b128(laddr, kg);
      async_copy_b128(laddr + 16, kg + 8);
    }
    // --- V chunk: staged transposed through VGPRs
    {
      const _Float16* vg = vbase + (size_t)(j0 + skl) * A_ + sa0;
      v8h v0 = *(const v8h*)(vg);
      v8h v1 = *(const v8h*)(vg + 8);
#pragma unroll
      for (int i = 0; i < 8; ++i) {
        Vt[sa0 + i][skl]     = v0[i];
        Vt[sa0 + 8 + i][skl] = v1[i];
      }
    }
    wait_async0();
    __syncthreads();

    // S = Q K^T for this wave's 16 rows x 64 kv cols (four 16x16 C tiles)
    v8f sacc[4];
#pragma unroll
    for (int t = 0; t < 4; ++t) {
      sacc[t] = zero8f();
#pragma unroll
      for (int ka = 0; ka < 2; ++ka) {
        const _Float16* kl = &Kt[lrow + 16 * t][ka * 32];  // lane's kv column
        v8h blo = *(const v8h*)(kl + khalf * 8);
        v8h bhi = *(const v8h*)(kl + 16 + khalf * 8);
        v16h bfrag = combine16(blo, bhi);
        sacc[t] = __builtin_amdgcn_wmma_f32_16x16x32_f16(
            false, qf[ka], false, bfrag, (short)0, sacc[t], false, false);
      }
    }

    // online softmax: row stats replicated across the 16 lanes of a half
    float p[4][8];
#pragma unroll
    for (int r = 0; r < 8; ++r) {
      float mx = fmaxf(fmaxf(sacc[0][r], sacc[1][r]), fmaxf(sacc[2][r], sacc[3][r]));
#pragma unroll
      for (int off = 8; off >= 1; off >>= 1)
        mx = fmaxf(mx, __shfl_xor(mx, off, 32));   // xor<16 stays within half
      const float newm = fmaxf(mrow[r], mx);
      const float corr = __expf(mrow[r] - newm);
      float rs = 0.f;
#pragma unroll
      for (int t = 0; t < 4; ++t) {
        p[t][r] = __expf(sacc[t][r] - newm);
        rs += p[t][r];
      }
#pragma unroll
      for (int off = 8; off >= 1; off >>= 1)
        rs += __shfl_xor(rs, off, 32);
      lsum[r] = lsum[r] * corr + rs;
      mrow[r] = newm;
#pragma unroll
      for (int t = 0; t < 4; ++t) acc[t][r] *= corr;
    }

    // relayout P: C-fragment -> A-fragment through wave-private LDS
    _Float16* pw = &Pl[wave][0];
#pragma unroll
    for (int t = 0; t < 4; ++t)
#pragma unroll
      for (int r = 0; r < 8; ++r)
        pw[(r + 8 * khalf) * 64 + lrow + 16 * t] = (_Float16)p[t][r];
    asm volatile("" ::: "memory");   // keep DS store->load order (HW DS is in-order per wave)
    v16h pfrag[2];
#pragma unroll
    for (int kc = 0; kc < 2; ++kc) {
      const _Float16* pr = pw + lrow * 64 + kc * 32;
      v8h plo = *(const v8h*)(pr + khalf * 8);
      v8h phi = *(const v8h*)(pr + 16 + khalf * 8);
      pfrag[kc] = combine16(plo, phi);
    }

    // acc += P @ V  (K = 64 kv, N = 64 a)
#pragma unroll
    for (int t = 0; t < 4; ++t) {
#pragma unroll
      for (int kc = 0; kc < 2; ++kc) {
        const _Float16* vl = &Vt[lrow + 16 * t][kc * 32];   // lane's output column a
        v8h blo = *(const v8h*)(vl + khalf * 8);
        v8h bhi = *(const v8h*)(vl + 16 + khalf * 8);
        v16h bfrag = combine16(blo, bhi);
        acc[t] = __builtin_amdgcn_wmma_f32_16x16x32_f16(
            false, pfrag[kc], false, bfrag, (short)0, acc[t], false, false);
      }
    }
    __syncthreads();
  }

  // epilogue: out[b][s][h*A + a] = acc / l
  float* obase = out + ((size_t)b * S_ + (size_t)(s0 + wave * 16)) * (H_ * A_) + h * A_;
#pragma unroll
  for (int t = 0; t < 4; ++t) {
    const int n = lrow + 16 * t;
#pragma unroll
    for (int r = 0; r < 8; ++r) {
      const int row = r + 8 * khalf;
      obase[(size_t)row * (H_ * A_) + n] = acc[t][r] / lsum[r];
    }
  }
}

// ---------------------------------------------------------------------------
extern "C" void kernel_launch(void* const* d_in, const int* in_sizes, int n_in,
                              void* d_out, int out_size, void* d_ws, size_t ws_size,
                              hipStream_t stream) {
  (void)in_sizes; (void)n_in; (void)out_size; (void)ws_size;
  const float* q  = (const float*)d_in[0];
  const float* k  = (const float*)d_in[1];
  const float* v  = (const float*)d_in[2];
  const float* Wq = (const float*)d_in[3];
  const float* bq = (const float*)d_in[4];
  const float* Wk = (const float*)d_in[5];
  const float* bk = (const float*)d_in[6];
  const float* Wv = (const float*)d_in[7];
  const float* bv = (const float*)d_in[8];

  const size_t headElems = (size_t)B_ * H_ * S_ * A_;   // 8.4M f16 each
  _Float16* qh = (_Float16*)d_ws;
  _Float16* kh = qh + headElems;
  _Float16* vh = kh + headElems;

  dim3 gProj(S_ / 128, B_ * H_, 3);
  proj_kernel<<<gProj, 256, 0, stream>>>(q, k, v, Wq, Wk, Wv, bq, bk, bv, qh, kh, vh);

  dim3 gAttn(S_ / 128, B_ * H_);
  attn_kernel<<<gAttn, 256, 0, stream>>>(qh, kh, vh, (float*)d_out);
}
